// CrossViewSwapAttention_661424963680
// MI455X (gfx1250) — matrix operand order repaired and checked by reading, not compiled, not run on hardware
//
#include <hip/hip_runtime.h>
#include <hip/hip_bf16.h>

typedef __attribute__((ext_vector_type(16))) _Float16 v16h;
typedef __attribute__((ext_vector_type(8)))  float    v8f;

#define LN_EPS 1e-5f
#define BN_EPS 1e-5f

union FragAB { v16h v; _Float16 h[16]; uint4 q[2]; };

// A-matrix fragment (16x32 f16), A row-major [rows][lda] in LDS.
// lanes 0-15: row=lane,    halves = K{k0+0..7, k0+16..23}
// lanes16-31: row=lane-16, halves = K{k0+8..15, k0+24..31}
__device__ __forceinline__ v16h ld_frag_a(const _Float16* A, int lda, int row0, int k0, int lane) {
  int r  = lane & 15;
  int kb = (lane >> 4) << 3;                    // 0 or 8
  const _Float16* p = A + (row0 + r) * lda + k0 + kb;
  FragAB f;
  f.q[0] = *(const uint4*)(p);
  f.q[1] = *(const uint4*)(p + 16);
  return f.v;
}

// B-matrix fragment (32x16 f16) with B[k][n] = W[n][k], W row-major [N][ldk] in LDS.
// lanes 0-15: col=lane,    K{k0+0..7, k0+16..23} ; lanes 16-31: col=lane-16, K{k0+8..15, k0+24..31}
__device__ __forceinline__ v16h ld_frag_b(const _Float16* W, int ldk, int n0, int k0, int lane) {
  int n  = lane & 15;
  int kb = (lane >> 4) << 3;                    // 0 or 8
  const _Float16* p = W + (n0 + n) * ldk + k0 + kb;
  FragAB f;
  f.q[0] = *(const uint4*)(p);
  f.q[1] = *(const uint4*)(p + 16);
  return f.v;
}

__device__ __forceinline__ v8f wmma32(v16h a, v16h b, v8f c) {
  return __builtin_amdgcn_wmma_f32_16x16x32_f16(false, a, false, b, (short)0, c, false, false);
}

// CDNA5 async global->LDS 16-byte copy (GV mode, no VGPR round trip; ASYNCcnt).
// Generic pointers to LDS have the wave-relative LDS byte offset in addr[31:0].
__device__ __forceinline__ void async_copy16(const void* gsrc, void* ldst) {
  unsigned lds = (unsigned)(unsigned long long)ldst;
  unsigned long long ga = (unsigned long long)gsrc;
  asm volatile("global_load_async_to_lds_b128 %0, %1, off" :: "v"(lds), "v"(ga) : "memory");
}
__device__ __forceinline__ void async_wait0() {
  asm volatile("s_wait_asynccnt 0x0" ::: "memory");
}

// stage a pre-converted f16 weight matrix of `halves` elements into LDS
__device__ __forceinline__ void stage_w_async(const _Float16* __restrict__ g,
                                              _Float16* l, int halves, int tid) {
  int chunks = halves >> 3;                     // 16B chunks
  for (int u = tid; u < chunks; u += 256)
    async_copy16(g + u * 8, l + u * 8);
  async_wait0();
}

// ---------------------------------------------------------------- prologue
__global__ void wcvt_kernel(const float* __restrict__ src, _Float16* __restrict__ dst) {
  int i = blockIdx.x * 256 + threadIdx.x;
  dst[i] = (_Float16)src[i];
}

__global__ void cembed_kernel(const float* __restrict__ cam_w,
                              const float* __restrict__ E_inv,
                              float* __restrict__ cemb) {
  int o = threadIdx.x;                               // 128 threads
  for (int v = 0; v < 4; ++v) {
    float a = 0.f;
    for (int i = 0; i < 4; ++i) a += cam_w[o * 4 + i] * E_inv[v * 16 + i * 4 + 3];
    cemb[v * 128 + o] = a;
  }
}

__global__ void imgembed_kernel(const float* __restrict__ I_inv,
                                const float* __restrict__ E_inv,
                                const float* __restrict__ img_w,
                                const float* __restrict__ cemb,
                                float* __restrict__ ie) {
  int gid = blockIdx.x * blockDim.x + threadIdx.x;   // 4096
  int v = gid >> 10, p = gid & 1023;
  int pi = p >> 5, pj = p & 31;
  float px = 256.0f * (float)pj / 31.0f;
  float py = 256.0f * (float)pi / 31.0f;
  float cam[3];
  for (int r = 0; r < 3; ++r)
    cam[r] = I_inv[v * 9 + r * 3 + 0] * px + I_inv[v * 9 + r * 3 + 1] * py + I_inv[v * 9 + r * 3 + 2];
  float dv[4];
  for (int r = 0; r < 4; ++r)
    dv[r] = E_inv[v * 16 + r * 4 + 0] * cam[0] + E_inv[v * 16 + r * 4 + 1] * cam[1] +
            E_inv[v * 16 + r * 4 + 2] * cam[2] + E_inv[v * 16 + r * 4 + 3];
  float ss = 0.f;
  for (int o = 0; o < 128; ++o) {
    float e = img_w[o * 4 + 0] * dv[0] + img_w[o * 4 + 1] * dv[1] +
              img_w[o * 4 + 2] * dv[2] + img_w[o * 4 + 3] * dv[3] - cemb[v * 128 + o];
    ss += e * e;
  }
  float rn = 1.0f / (sqrtf(ss) + 1e-7f);
  for (int o = 0; o < 128; ++o) {
    float e = img_w[o * 4 + 0] * dv[0] + img_w[o * 4 + 1] * dv[1] +
              img_w[o * 4 + 2] * dv[2] + img_w[o * 4 + 3] * dv[3] - cemb[v * 128 + o];
    ie[(size_t)(v * 128 + o) * 1024 + p] = e * rn;
  }
}

__global__ __launch_bounds__(128) void keyval_kernel(
    const float* __restrict__ feature, const float* __restrict__ ie,
    const float* __restrict__ fpg, const float* __restrict__ fpb,
    const float* __restrict__ fpm, const float* __restrict__ fpv, const float* __restrict__ fpw,
    const float* __restrict__ flg, const float* __restrict__ flb,
    const float* __restrict__ flm, const float* __restrict__ flv, const float* __restrict__ flw,
    float* __restrict__ keyB, float* __restrict__ valB) {
  __shared__ float aP[128], aL[128];
  int v = blockIdx.x >> 10, p = blockIdx.x & 1023, c = threadIdx.x;
  float f = feature[(size_t)(v * 128 + c) * 1024 + p];
  float xp = (f - fpm[c]) * rsqrtf(fpv[c] + BN_EPS) * fpg[c] + fpb[c];
  float xl = (f - flm[c]) * rsqrtf(flv[c] + BN_EPS) * flg[c] + flb[c];
  aP[c] = fmaxf(xp, 0.f);
  aL[c] = fmaxf(xl, 0.f);
  __syncthreads();
  float accK = 0.f, accV = 0.f;
  for (int i = 0; i < 128; ++i) {
    accK += fpw[c * 128 + i] * aP[i];
    accV += flw[c * 128 + i] * aL[i];
  }
  size_t gi = (size_t)(v * 128 + c) * 1024 + p;
  keyB[gi] = ie[gi] + accK;
  valB[gi] = accV;
}

__global__ void initz_kernel(const float* __restrict__ x, float* __restrict__ z) {
  int s = blockIdx.x, c = threadIdx.x;               // 65536 blocks x 128
  z[(size_t)s * 128 + c] = x[(size_t)c * 65536 + s];
}

// ------------------------------------------------ fused windowed attention
// grid: stage1 = 1024 blocks (view,window), stage2 = 256 blocks (window only,
// queries identical across views so one view with vscale=1 is exact).
// block = 256 threads (8 wave32). LDS (dynamic, 245760 B):
//   sQ  @0      : 256x128 f16  (LN'd q -> later attended heads; WMMA A)
//   sW  @65536  : 128x128 f16  (weight tile staged via async->LDS; WMMA B)
//   sS  @98304  : 256x128 f32  (k/v raw+norm scratch -> q raw -> QH)
//   sKH @229376 : 16x128 f32, sVH @237568 : 16x128 f32
__global__ __launch_bounds__(256) void attn_win_kernel(
    int stage, float vscale,
    const float* __restrict__ keyB, const float* __restrict__ valB,
    const float* __restrict__ xin,  const float* __restrict__ zin,
    const float* __restrict__ bev_w, const float* __restrict__ bev_b,
    const float* __restrict__ cemb,
    const float* __restrict__ nq_g, const float* __restrict__ nq_b,
    const _Float16* __restrict__ wqh, const float* __restrict__ bq,
    const float* __restrict__ nk_g, const float* __restrict__ nk_b,
    const _Float16* __restrict__ wkh, const float* __restrict__ bk,
    const float* __restrict__ nv_g, const float* __restrict__ nv_b,
    const _Float16* __restrict__ wvh, const float* __restrict__ bv,
    const _Float16* __restrict__ wph, const float* __restrict__ bp,
    float* __restrict__ zout) {
  extern __shared__ char smem[];
  _Float16* sQ  = (_Float16*)(smem);
  _Float16* sW  = (_Float16*)(smem + 65536);
  float*    sS  = (float*)(smem + 98304);
  float*    sKH = (float*)(smem + 229376);
  float*    sVH = (float*)(smem + 237568);

  const int tid  = threadIdx.x;
  const int lane = tid & 31;
  const int wave = tid >> 5;
  const int wIdx = blockIdx.x & 255;
  const int view = blockIdx.x >> 8;
  const int xw = wIdx >> 4, yw = wIdx & 15;

  // ---- P1: gather 16 k/v tokens (all views), LN, project to heads ----
  for (int u = tid; u < 2048; u += 256) {
    int tok = u >> 7, c = u & 127;
    int kv = tok >> 2, w1 = (tok >> 1) & 1, w2 = tok & 1;
    int fi, fj;
    if (stage == 1) { fi = xw * 2 + w1;  fj = yw * 2 + w2;  }   // local windows
    else            { fi = w1 * 16 + xw; fj = w2 * 16 + yw; }   // grid windows
    size_t gi = (size_t)(kv * 128 + c) * 1024 + fi * 32 + fj;
    sS[u]        = keyB[gi];
    sS[2048 + u] = valB[gi];
  }
  stage_w_async(wkh, sW, 16384, tid);                // overlap: async wk stage
  __syncthreads();
  if (tid < 32) {                                    // LN over 128ch of 32 tokens
    int isV = tid >> 4, t = tid & 15;
    float* row = sS + isV * 2048 + t * 128;
    float s = 0.f, s2 = 0.f;
    for (int c = 0; c < 128; ++c) { float x = row[c]; s += x; s2 += x * x; }
    float m = s * (1.f / 128.f);
    float var = s2 * (1.f / 128.f) - m * m;
    float rs = rsqrtf(var + LN_EPS);
    const float* g = isV ? nv_g : nk_g;
    const float* b = isV ? nv_b : nk_b;
    for (int c = 0; c < 128; ++c) row[c] = (row[c] - m) * rs * g[c] + b[c];
  }
  __syncthreads();
  for (int u = tid; u < 2048; u += 256) {            // K projection (tiny, VALU)
    int tok = u >> 7, o = u & 127;
    float acc = bk[o];
    const float* kn = sS + tok * 128;
    const _Float16* wr = sW + o * 128;
    for (int i = 0; i < 128; ++i) acc += (float)wr[i] * kn[i];
    sKH[u] = acc;
  }
  __syncthreads();
  stage_w_async(wvh, sW, 16384, tid);
  __syncthreads();
  for (int u = tid; u < 2048; u += 256) {            // V projection
    int tok = u >> 7, o = u & 127;
    float acc = bv[o];
    const float* vn = sS + 2048 + tok * 128;
    const _Float16* wr = sW + o * 128;
    for (int i = 0; i < 128; ++i) acc += (float)wr[i] * vn[i];
    sVH[u] = acc;
  }
  __syncthreads();

  // ---- P2: build 256 q tokens on the fly, LN -> f16 A-matrix ----
  {
    int t = tid;
    int w1 = t >> 4, w2 = t & 15;
    int i = xw * 16 + w1, j = yw * 16 + w2;
    int sp = i * 256 + j;
    float* qrow = sS + t * 128;
    float s = 0.f, s2 = 0.f;
    if (stage == 1) {
      // analytic BEV world grid: world0=(256*i/255-128)/(-2.56), world1=(256*j/255-128)/(-2.56)
      float wx = (256.0f * (float)i / 255.0f - 128.0f) * (-100.0f / 256.0f);
      float wy = (256.0f * (float)j / 255.0f - 128.0f) * (-100.0f / 256.0f);
      float ss = 0.f;
      for (int c = 0; c < 128; ++c) {
        float e = bev_w[2 * c] * wx + bev_w[2 * c + 1] * wy + bev_b[c] - cemb[view * 128 + c];
        ss += e * e;
      }
      float rn = 1.0f / (sqrtf(ss) + 1e-7f);
      for (int c = 0; c < 128; ++c) {
        float e = bev_w[2 * c] * wx + bev_w[2 * c + 1] * wy + bev_b[c] - cemb[view * 128 + c];
        float q = e * rn + xin[(size_t)c * 65536 + sp];
        qrow[c] = q; s += q; s2 += q * q;
      }
    } else {
      for (int c = 0; c < 128; ++c) {
        float q = zin[(size_t)sp * 128 + c];
        qrow[c] = q; s += q; s2 += q * q;
      }
    }
    float m = s * (1.f / 128.f);
    float var = s2 * (1.f / 128.f) - m * m;
    float rs = rsqrtf(var + LN_EPS);
    for (int c = 0; c < 128; ++c)
      sQ[t * 128 + c] = (_Float16)((qrow[c] - m) * rs * nq_g[c] + nq_b[c]);
  }
  stage_w_async(wqh, sW, 16384, tid);
  __syncthreads();

  // ---- P3: QH(256x128) = Qn x Wq^T + bq  (WMMA, wave owns one N-tile) ----
  {
    v16h b0 = ld_frag_b(sW, 128, wave * 16, 0,  lane);
    v16h b1 = ld_frag_b(sW, 128, wave * 16, 32, lane);
    v16h b2 = ld_frag_b(sW, 128, wave * 16, 64, lane);
    v16h b3 = ld_frag_b(sW, 128, wave * 16, 96, lane);
    int col = wave * 16 + (lane & 15);
    float bqv = bq[col];
    int rowoff = (lane >> 4) << 3;
    for (int mt = 0; mt < 16; ++mt) {
      v8f acc = {};
      acc = wmma32(ld_frag_a(sQ, 128, mt * 16, 0,  lane), b0, acc);
      acc = wmma32(ld_frag_a(sQ, 128, mt * 16, 32, lane), b1, acc);
      acc = wmma32(ld_frag_a(sQ, 128, mt * 16, 64, lane), b2, acc);
      acc = wmma32(ld_frag_a(sQ, 128, mt * 16, 96, lane), b3, acc);
#pragma unroll
      for (int r = 0; r < 8; ++r) {
        int tok = mt * 16 + r + rowoff;
        sS[tok * 128 + col] = acc[r] + bqv;
      }
    }
  }
  __syncthreads();

  // ---- P4: 16-key attention per token (4 heads x 32) ----
  {
    int t = tid;
    const float* qh = sS + t * 128;
#pragma unroll
    for (int h = 0; h < 4; ++h) {
      int base = h * 32;
      float sc[16];
      float mx = -1e30f;
#pragma unroll
      for (int k = 0; k < 16; ++k) {
        float d = 0.f;
        const float* kr = sKH + k * 128 + base;
#pragma unroll
        for (int dd = 0; dd < 32; ++dd) d += qh[base + dd] * kr[dd];
        d *= 0.17677669529663687f;                   // 32^-0.5
        sc[k] = d; mx = fmaxf(mx, d);
      }
      float sum = 0.f;
#pragma unroll
      for (int k = 0; k < 16; ++k) { sc[k] = expf(sc[k] - mx); sum += sc[k]; }
      float rsum = 1.0f / sum;
      for (int dd = 0; dd < 32; ++dd) {
        float a = 0.f;
#pragma unroll
        for (int k = 0; k < 16; ++k) a += sc[k] * sVH[k * 128 + base + dd];
        sQ[t * 128 + base + dd] = (_Float16)(a * rsum);
      }
    }
  }
  stage_w_async(wph, sW, 16384, tid);
  __syncthreads();

  // ---- P5: PROJ = Attn x Wp^T; view-mean accumulate (+bp) into zout ----
  {
    v16h b0 = ld_frag_b(sW, 128, wave * 16, 0,  lane);
    v16h b1 = ld_frag_b(sW, 128, wave * 16, 32, lane);
    v16h b2 = ld_frag_b(sW, 128, wave * 16, 64, lane);
    v16h b3 = ld_frag_b(sW, 128, wave * 16, 96, lane);
    int col = wave * 16 + (lane & 15);
    float bpv = bp[col];
    int rowoff = (lane >> 4) << 3;
    for (int mt = 0; mt < 16; ++mt) {
      v8f acc = {};
      acc = wmma32(ld_frag_a(sQ, 128, mt * 16, 0,  lane), b0, acc);
      acc = wmma32(ld_frag_a(sQ, 128, mt * 16, 32, lane), b1, acc);
      acc = wmma32(ld_frag_a(sQ, 128, mt * 16, 64, lane), b2, acc);
      acc = wmma32(ld_frag_a(sQ, 128, mt * 16, 96, lane), b3, acc);
#pragma unroll
      for (int r = 0; r < 8; ++r) {
        int tok = mt * 16 + r + rowoff;
        int w1 = tok >> 4, w2 = tok & 15;
        int sp = (xw * 16 + w1) * 256 + (yw * 16 + w2);
        atomicAdd(&zout[(size_t)sp * 128 + col], vscale * (acc[r] + bpv));
      }
    }
  }
}

// ------------------------------------------------ residual MLP (LN->W1->GELU->W2)+x
// block = 256 threads, 128 tokens. LDS 163840 B: sA1 32K | sA2 64K | sWt 64K.
__global__ __launch_bounds__(256) void mlp_kernel(
    float* __restrict__ z,
    const float* __restrict__ g,  const float* __restrict__ b,
    const _Float16* __restrict__ w1h, const float* __restrict__ b1,
    const _Float16* __restrict__ w2h, const float* __restrict__ b2) {
  extern __shared__ char smem[];
  _Float16* sA1 = (_Float16*)(smem);
  _Float16* sA2 = (_Float16*)(smem + 32768);
  _Float16* sWt = (_Float16*)(smem + 98304);
  const int tid = threadIdx.x, lane = tid & 31, wave = tid >> 5;
  const int s0 = blockIdx.x * 128;

  stage_w_async(w1h, sWt, 32768, tid);               // async W1 stage first
  if (tid < 128) {
    const float* row = z + (size_t)(s0 + tid) * 128;
    float s = 0.f, s2 = 0.f;
    for (int c = 0; c < 128; ++c) { float x = row[c]; s += x; s2 += x * x; }
    float m = s * (1.f / 128.f);
    float var = s2 * (1.f / 128.f) - m * m;
    float rs = rsqrtf(var + LN_EPS);
    for (int c = 0; c < 128; ++c) sA1[tid * 128 + c] = (_Float16)((row[c] - m) * rs * g[c] + b[c]);
  }
  __syncthreads();

  // GEMM1: H(128x256) = A1(128x128) * W1^T ; exact GELU
  for (int nt = wave; nt < 16; nt += 8) {
    v16h f0 = ld_frag_b(sWt, 128, nt * 16, 0,  lane);
    v16h f1 = ld_frag_b(sWt, 128, nt * 16, 32, lane);
    v16h f2 = ld_frag_b(sWt, 128, nt * 16, 64, lane);
    v16h f3 = ld_frag_b(sWt, 128, nt * 16, 96, lane);
    int col = nt * 16 + (lane & 15);
    float bv = b1[col];
    int rowoff = (lane >> 4) << 3;
    for (int mt = 0; mt < 8; ++mt) {
      v8f acc = {};
      acc = wmma32(ld_frag_a(sA1, 128, mt * 16, 0,  lane), f0, acc);
      acc = wmma32(ld_frag_a(sA1, 128, mt * 16, 32, lane), f1, acc);
      acc = wmma32(ld_frag_a(sA1, 128, mt * 16, 64, lane), f2, acc);
      acc = wmma32(ld_frag_a(sA1, 128, mt * 16, 96, lane), f3, acc);
#pragma unroll
      for (int r = 0; r < 8; ++r) {
        int tok = mt * 16 + r + rowoff;
        float h = acc[r] + bv;
        float ge = 0.5f * h * (1.0f + erff(h * 0.7071067811865475f));
        sA2[tok * 256 + col] = (_Float16)ge;
      }
    }
  }
  __syncthreads();
  stage_w_async(w2h, sWt, 32768, tid);
  __syncthreads();

  // GEMM2: OUT(128x128) = A2(128x256) * W2^T ; residual into z
  {
    int nt = wave;
    v16h bfr[8];
#pragma unroll
    for (int ks = 0; ks < 8; ++ks) bfr[ks] = ld_frag_b(sWt, 256, nt * 16, ks * 32, lane);
    int col = nt * 16 + (lane & 15);
    float bv = b2[col];
    int rowoff = (lane >> 4) << 3;
    for (int mt = 0; mt < 8; ++mt) {
      v8f acc = {};
#pragma unroll
      for (int ks = 0; ks < 8; ++ks)
        acc = wmma32(ld_frag_a(sA2, 256, mt * 16, ks * 32, lane), bfr[ks], acc);
#pragma unroll
      for (int r = 0; r < 8; ++r) {
        int tok = mt * 16 + r + rowoff;
        size_t gi = (size_t)(s0 + tok) * 128 + col;
        z[gi] = z[gi] + acc[r] + bv;
      }
    }
  }
}

__global__ __launch_bounds__(256) void post_kernel(const float* __restrict__ z,
                                                   const float* __restrict__ g,
                                                   const float* __restrict__ b,
                                                   float* __restrict__ out) {
  int s = blockIdx.x * 256 + threadIdx.x;
  const float* row = z + (size_t)s * 128;
  float su = 0.f, s2 = 0.f;
  for (int c = 0; c < 128; ++c) { float x = row[c]; su += x; s2 += x * x; }
  float m = su * (1.f / 128.f);
  float var = s2 * (1.f / 128.f) - m * m;
  float rs = rsqrtf(var + LN_EPS);
  for (int c = 0; c < 128; ++c)
    out[(size_t)c * 65536 + s] = (row[c] - m) * rs * g[c] + b[c];
}

// ---------------------------------------------------------------- launch
extern "C" void kernel_launch(void* const* d_in, const int* in_sizes, int n_in,
                              void* d_out, int out_size, void* d_ws, size_t ws_size,
                              hipStream_t stream) {
  const float* x       = (const float*)d_in[1];
  const float* feature = (const float*)d_in[2];
  const float* I_inv   = (const float*)d_in[3];
  const float* E_inv   = (const float*)d_in[4];
  const float* fl_g = (const float*)d_in[5],  *fl_b = (const float*)d_in[6];
  const float* fl_m = (const float*)d_in[7],  *fl_v = (const float*)d_in[8];
  const float* fl_w = (const float*)d_in[9];
  const float* fp_g = (const float*)d_in[10], *fp_b = (const float*)d_in[11];
  const float* fp_m = (const float*)d_in[12], *fp_v = (const float*)d_in[13];
  const float* fp_w = (const float*)d_in[14];
  const float* bev_w = (const float*)d_in[15], *bev_b = (const float*)d_in[16];
  const float* img_w = (const float*)d_in[17], *cam_w = (const float*)d_in[18];
  const float* a1_nq_g = (const float*)d_in[19], *a1_nq_b = (const float*)d_in[20];
  const float* a1_wq = (const float*)d_in[21],   *a1_bq = (const float*)d_in[22];
  const float* a1_nk_g = (const float*)d_in[23], *a1_nk_b = (const float*)d_in[24];
  const float* a1_wk = (const float*)d_in[25],   *a1_bk = (const float*)d_in[26];
  const float* a1_nv_g = (const float*)d_in[27], *a1_nv_b = (const float*)d_in[28];
  const float* a1_wv = (const float*)d_in[29],   *a1_bv = (const float*)d_in[30];
  const float* a1_wp = (const float*)d_in[31],   *a1_bp = (const float*)d_in[32];
  const float* pn1_g = (const float*)d_in[33],   *pn1_b = (const float*)d_in[34];
  const float* m1_w1 = (const float*)d_in[35],   *m1_b1 = (const float*)d_in[36];
  const float* m1_w2 = (const float*)d_in[37],   *m1_b2 = (const float*)d_in[38];
  const float* a2_nq_g = (const float*)d_in[39], *a2_nq_b = (const float*)d_in[40];
  const float* a2_wq = (const float*)d_in[41],   *a2_bq = (const float*)d_in[42];
  const float* a2_nk_g = (const float*)d_in[43], *a2_nk_b = (const float*)d_in[44];
  const float* a2_wk = (const float*)d_in[45],   *a2_bk = (const float*)d_in[46];
  const float* a2_nv_g = (const float*)d_in[47], *a2_nv_b = (const float*)d_in[48];
  const float* a2_wv = (const float*)d_in[49],   *a2_bv = (const float*)d_in[50];
  const float* a2_wp = (const float*)d_in[51],   *a2_bp = (const float*)d_in[52];
  const float* pn2_g = (const float*)d_in[53],   *pn2_b = (const float*)d_in[54];
  const float* m2_w1 = (const float*)d_in[55],   *m2_b1 = (const float*)d_in[56];
  const float* m2_w2 = (const float*)d_in[57],   *m2_b2 = (const float*)d_in[58];
  const float* post_g = (const float*)d_in[59],  *post_b = (const float*)d_in[60];

  float* ws   = (float*)d_ws;
  float* cemb = ws;                       // 512 (+pad)
  float* ie   = ws + 1024;                // 4*128*1024
  float* keyB = ie + 524288;
  float* valB = keyB + 524288;
  float* zA   = valB + 524288;            // 65536*128
  float* zB   = zA + 8388608;
  _Float16* wh = (_Float16*)(zB + 8388608);   // 12 pre-converted f16 weights
  _Float16* h_a1wq = wh;            _Float16* h_a1wk = wh + 16384;
  _Float16* h_a1wv = wh + 32768;    _Float16* h_a1wp = wh + 49152;
  _Float16* h_a2wq = wh + 65536;    _Float16* h_a2wk = wh + 81920;
  _Float16* h_a2wv = wh + 98304;    _Float16* h_a2wp = wh + 114688;
  _Float16* h_m1w1 = wh + 131072;   _Float16* h_m1w2 = wh + 163840;
  _Float16* h_m2w1 = wh + 196608;   _Float16* h_m2w2 = wh + 229376;

  // one-time f32 -> f16 weight conversion (64 blocks = 16384 elems, 128 = 32768)
  wcvt_kernel<<<64, 256, 0, stream>>>(a1_wq, h_a1wq);
  wcvt_kernel<<<64, 256, 0, stream>>>(a1_wk, h_a1wk);
  wcvt_kernel<<<64, 256, 0, stream>>>(a1_wv, h_a1wv);
  wcvt_kernel<<<64, 256, 0, stream>>>(a1_wp, h_a1wp);
  wcvt_kernel<<<64, 256, 0, stream>>>(a2_wq, h_a2wq);
  wcvt_kernel<<<64, 256, 0, stream>>>(a2_wk, h_a2wk);
  wcvt_kernel<<<64, 256, 0, stream>>>(a2_wv, h_a2wv);
  wcvt_kernel<<<64, 256, 0, stream>>>(a2_wp, h_a2wp);
  wcvt_kernel<<<128, 256, 0, stream>>>(m1_w1, h_m1w1);
  wcvt_kernel<<<128, 256, 0, stream>>>(m1_w2, h_m1w2);
  wcvt_kernel<<<128, 256, 0, stream>>>(m2_w1, h_m2w1);
  wcvt_kernel<<<128, 256, 0, stream>>>(m2_w2, h_m2w2);

  cembed_kernel<<<1, 128, 0, stream>>>(cam_w, E_inv, cemb);
  imgembed_kernel<<<16, 256, 0, stream>>>(I_inv, E_inv, img_w, cemb, ie);
  keyval_kernel<<<4096, 128, 0, stream>>>(feature, ie,
      fp_g, fp_b, fp_m, fp_v, fp_w, fl_g, fl_b, fl_m, fl_v, fl_w, keyB, valB);
  initz_kernel<<<65536, 128, 0, stream>>>(x, zA);      // zA = skip1 (x in HWC)

  const size_t attnLds = 245760;
  attn_win_kernel<<<1024, 256, attnLds, stream>>>(1, 0.25f, keyB, valB, x, zA,
      bev_w, bev_b, cemb,
      a1_nq_g, a1_nq_b, h_a1wq, a1_bq, a1_nk_g, a1_nk_b, h_a1wk, a1_bk,
      a1_nv_g, a1_nv_b, h_a1wv, a1_bv, h_a1wp, a1_bp, zA);
  mlp_kernel<<<512, 256, 163840, stream>>>(zA, pn1_g, pn1_b, h_m1w1, m1_b1, h_m1w2, m1_b2);

  hipMemcpyAsync(zB, zA, (size_t)8388608 * sizeof(float), hipMemcpyDeviceToDevice, stream);
  // stage 2: queries are broadcast across views -> per-view outputs identical;
  // run a single view with scale 1.0 (exactly equals the 4-view mean).
  attn_win_kernel<<<256, 256, attnLds, stream>>>(2, 1.0f, keyB, valB, x, zA,
      bev_w, bev_b, cemb,
      a2_nq_g, a2_nq_b, h_a2wq, a2_bq, a2_nk_g, a2_nk_b, h_a2wk, a2_bk,
      a2_nv_g, a2_nv_b, h_a2wv, a2_bv, h_a2wp, a2_bp, zB);
  mlp_kernel<<<512, 256, 163840, stream>>>(zB, pn2_g, pn2_b, h_m2w1, m2_b1, h_m2w2, m2_b2);

  post_kernel<<<256, 256, 0, stream>>>(zB, post_g, post_b, (float*)d_out);
}
